// PlainEncoderLayer_47691316855184
// MI455X (gfx1250) — matrix operand order repaired and checked by reading, not compiled, not run on hardware
//
#include <hip/hip_runtime.h>

// ---------------------------------------------------------------------------
// Types for CDNA5 WMMA (wave32)
// ---------------------------------------------------------------------------
typedef __attribute__((ext_vector_type(16))) short  v16s;
typedef __attribute__((ext_vector_type(8)))  short  v8s;
typedef __attribute__((ext_vector_type(16))) __bf16 v16bf;
typedef __attribute__((ext_vector_type(8)))  float  v8f;

#define HIDDEN 768
#define HEADS  12
#define DHEAD  64
#define INTER  3072
#define BATCH  4
#define TSEQ   2048
#define MTOT   (BATCH * TSEQ)                  // 8192 rows
#define QSZ    (BATCH * HEADS * TSEQ * DHEAD)  // 6291456 elems per q/k/v buffer

// float -> bf16 round-to-nearest-even
static __device__ inline unsigned short f2bf(float f) {
  unsigned int u = __builtin_bit_cast(unsigned int, f);
  u += 0x7FFFu + ((u >> 16) & 1u);
  return (unsigned short)(u >> 16);
}

// A-fragment (16x32 bf16): per-lane elems = K[base..base+7] and K[base+16..base+23],
// base = (lane>>4)*8. p must point at row + kstep + (lane>>4)*8.
static __device__ inline v16bf make_afrag(const unsigned short* p) {
  v8s lo = *(const v8s*)p;
  v8s hi = *(const v8s*)(p + 16);
  v16s r = __builtin_shufflevector(lo, hi, 0,1,2,3,4,5,6,7,8,9,10,11,12,13,14,15);
  return __builtin_bit_cast(v16bf, r);
}

// B-fragment (32x16 bf16): per-lane elems = K[base..base+7] and K[base+16..base+23]
// for column lane%16, base = (lane>>4)*8. p points at col-major row + kstep + (lane>>4)*8.
static __device__ inline v16bf make_bfrag(const unsigned short* p) {
  v8s lo = *(const v8s*)p;
  v8s hi = *(const v8s*)(p + 16);
  v16s r = __builtin_shufflevector(lo, hi, 0,1,2,3,4,5,6,7,8,9,10,11,12,13,14,15);
  return __builtin_bit_cast(v16bf, r);
}

static __device__ inline v8f wmma_bf16(v16bf a, v16bf b, v8f c) {
  return __builtin_amdgcn_wmma_f32_16x16x32_bf16(false, a, false, b, (short)0, c,
                                                 false, false);
}

// Async copy 16 bytes global -> LDS (per lane), tracked by ASYNCcnt.
static __device__ inline void async_g2l_b128(unsigned ldsoff,
                                             const unsigned short* gp) {
  asm volatile("global_load_async_to_lds_b128 %0, %1, off"
               :
               : "v"(ldsoff), "v"(gp)
               : "memory");
}
static __device__ inline void wait_asynccnt0() {
  asm volatile("s_wait_asynccnt 0x0" ::: "memory");
}

// ---------------------------------------------------------------------------
// Weight transpose + fp32->bf16 convert: W[K,N] -> Wt[N,K]   (one-time)
// ---------------------------------------------------------------------------
__global__ __launch_bounds__(256) void convert_wT(const float* __restrict__ W,
                                                  unsigned short* __restrict__ Wt,
                                                  int K, int N) {
  __shared__ float tile[32][33];
  int bx = blockIdx.x;  // N / 32
  int by = blockIdx.y;  // K / 32
  int tx = threadIdx.x & 31, ty = threadIdx.x >> 5;  // 8 rows per pass
#pragma unroll
  for (int p = 0; p < 4; ++p) {
    int k = by * 32 + p * 8 + ty;
    tile[p * 8 + ty][tx] = W[(size_t)k * N + bx * 32 + tx];
  }
  __syncthreads();
#pragma unroll
  for (int p = 0; p < 4; ++p) {
    int n = bx * 32 + p * 8 + ty;
    Wt[(size_t)n * K + by * 32 + tx] = f2bf(tile[tx][p * 8 + ty]);
  }
}

// ---------------------------------------------------------------------------
// LayerNorm: one token (768 floats) per 256-thread block -> bf16 output
// ---------------------------------------------------------------------------
__global__ __launch_bounds__(256) void ln_kernel(const float* __restrict__ x,
                                                 const float* __restrict__ g,
                                                 const float* __restrict__ beta,
                                                 unsigned short* __restrict__ out) {
  int t = blockIdx.x;
  int tid = threadIdx.x;
  const float* xr = x + (size_t)t * HIDDEN;
  float a0 = xr[tid], a1 = xr[tid + 256], a2 = xr[tid + 512];
  __shared__ float red[256];
  red[tid] = a0 + a1 + a2;
  __syncthreads();
  for (int o = 128; o > 0; o >>= 1) {
    if (tid < o) red[tid] += red[tid + o];
    __syncthreads();
  }
  float mean = red[0] * (1.0f / 768.0f);
  __syncthreads();
  float d0 = a0 - mean, d1 = a1 - mean, d2 = a2 - mean;
  red[tid] = d0 * d0 + d1 * d1 + d2 * d2;
  __syncthreads();
  for (int o = 128; o > 0; o >>= 1) {
    if (tid < o) red[tid] += red[tid + o];
    __syncthreads();
  }
  float rinv = rsqrtf(red[0] * (1.0f / 768.0f) + 1e-5f);
  unsigned short* orow = out + (size_t)t * HIDDEN;
  orow[tid]       = f2bf(d0 * rinv * g[tid]       + beta[tid]);
  orow[tid + 256] = f2bf(d1 * rinv * g[tid + 256] + beta[tid + 256]);
  orow[tid + 512] = f2bf(d2 * rinv * g[tid + 512] + beta[tid + 512]);
}

// ---------------------------------------------------------------------------
// GEMM: C[8192,N] = A(bf16)[8192,K] x Bt(bf16)[N,K]^T
// Block tile 128x128, k-step 32. 8 waves: 4(M) x 2(N); wave tile 32x64.
// A tile staged to LDS via async global->LDS copy; B fragments direct global.
// EPI 0: QKV scatter (q scaled 1/8; q,k -> [B,H,T,D]; v -> [B,H,D,T])
// EPI 1: + residual extra -> outf (f32)
// EPI 2: + bias, poly-GELU -> outb (bf16)
// EPI 3: + bias + extra -> outf (f32)
// ---------------------------------------------------------------------------
template <int EPI>
__global__ __launch_bounds__(256) void gemm_kernel(
    const unsigned short* __restrict__ A, const unsigned short* __restrict__ Bt,
    int N, int K,
    const float* __restrict__ bias, const float* __restrict__ extra,
    const float* __restrict__ poly,
    unsigned short* __restrict__ outb, float* __restrict__ outf) {
  __shared__ __align__(64) unsigned short As[128 * 48];  // row-major, stride 48
  int tid = threadIdx.x;
  int lane = tid & 31, wid = tid >> 5;
  int waveM = wid & 3, waveN = wid >> 2;
  int lrow = lane & 15, lhalf = lane >> 4;
  int row0 = blockIdx.y * 128;
  int n0 = blockIdx.x * 128;

  v8f acc[2][4];
  v8f zero = {0.f, 0.f, 0.f, 0.f, 0.f, 0.f, 0.f, 0.f};
#pragma unroll
  for (int mt = 0; mt < 2; ++mt)
#pragma unroll
    for (int nt = 0; nt < 4; ++nt) acc[mt][nt] = zero;

  // A-tile copy mapping: 2 x B128 per thread (128 rows x 32 k of bf16)
  int r0c = tid >> 2, kk0 = (tid & 3) * 8;
  int r1c = (tid + 256) >> 2, kk1 = ((tid + 256) & 3) * 8;
  unsigned lds0 = (unsigned)(size_t)&As[r0c * 48 + kk0];
  unsigned lds1 = (unsigned)(size_t)&As[r1c * 48 + kk1];
  const unsigned short* ag0 = A + (size_t)(row0 + r0c) * K + kk0;
  const unsigned short* ag1 = A + (size_t)(row0 + r1c) * K + kk1;

  for (int k0 = 0; k0 < K; k0 += 32) {
    __syncthreads();
    async_g2l_b128(lds0, ag0 + k0);
    async_g2l_b128(lds1, ag1 + k0);
    wait_asynccnt0();
    __syncthreads();
    v16bf afr[2];
#pragma unroll
    for (int mt = 0; mt < 2; ++mt)
      afr[mt] = make_afrag(&As[(waveM * 32 + mt * 16 + lrow) * 48 + lhalf * 8]);
#pragma unroll
    for (int nt = 0; nt < 4; ++nt) {
      v16bf bfr = make_bfrag(Bt + (size_t)(n0 + waveN * 64 + nt * 16 + lrow) * K +
                             k0 + lhalf * 8);
#pragma unroll
      for (int mt = 0; mt < 2; ++mt)
        acc[mt][nt] = wmma_bf16(afr[mt], bfr, acc[mt][nt]);
    }
  }

  // epilogue: C layout row = vgpr + 8*(lane>>4), col = lane%16
#pragma unroll
  for (int mt = 0; mt < 2; ++mt) {
#pragma unroll
    for (int nt = 0; nt < 4; ++nt) {
      int col = n0 + waveN * 64 + nt * 16 + lrow;
      int rbase = row0 + waveM * 32 + mt * 16 + lhalf * 8;
#pragma unroll
      for (int v = 0; v < 8; ++v) {
        int row = rbase + v;
        float val = acc[mt][nt][v];
        if (EPI == 0) {
          int which = col / HIDDEN;
          int rem = col - which * HIDDEN;
          int head = rem >> 6, d = rem & 63;
          int bb = row >> 11, tq = row & 2047;
          size_t idx = ((size_t)(bb * HEADS + head) * TSEQ + tq) * DHEAD + d;
          if (which == 0)
            outb[idx] = f2bf(val * 0.125f);  // fold 1/sqrt(D) into q
          else if (which == 1)
            outb[(size_t)QSZ + idx] = f2bf(val);
          else
            outb[2 * (size_t)QSZ +
                 ((size_t)(bb * HEADS + head) * DHEAD + d) * TSEQ + tq] =
                f2bf(val);
        } else if (EPI == 1) {
          size_t idx = (size_t)row * HIDDEN + col;
          outf[idx] = val + extra[idx];
        } else if (EPI == 2) {
          float t = val + bias[col];
          outb[(size_t)row * INTER + col] =
              f2bf(poly[0] + poly[1] * t + poly[2] * t * t);
        } else {
          size_t idx = (size_t)row * HIDDEN + col;
          outf[idx] = val + bias[col] + extra[idx];
        }
      }
    }
  }
}

// ---------------------------------------------------------------------------
// Attention: softmax-free quadratic scores, normalized by row-sum.
// Block = 64 query rows of one (b,h). S = q k^T (q pre-scaled), a = (s+1)^2 * m.
// out = a @ [V | 1] accumulated over key tiles; last column is the denominator.
// ---------------------------------------------------------------------------
__global__ __launch_bounds__(256) void attn_kernel(
    const unsigned short* __restrict__ qb, const unsigned short* __restrict__ kb,
    const unsigned short* __restrict__ vtb, const int* __restrict__ maskp,
    unsigned short* __restrict__ ao) {
  __shared__ __align__(64) unsigned short Asc[64 * 80];  // scores bf16, stride 80
  __shared__ float dens[64];
  int tid = threadIdx.x;
  int lane = tid & 31, wid = tid >> 5;
  int waveM = wid & 3, waveN = wid >> 2;
  int lrow = lane & 15, lhalf = lane >> 4;
  int bh = blockIdx.y;
  int b = bh / HEADS, hh = bh % HEADS;
  int q0 = blockIdx.x * 64;
  const unsigned short* qbase = qb + (size_t)bh * TSEQ * DHEAD;
  const unsigned short* kbase = kb + (size_t)bh * TSEQ * DHEAD;
  const unsigned short* vbase = vtb + (size_t)bh * DHEAD * TSEQ;

  // Q fragments for this wave's 16 rows (loaded once)
  int qrow = q0 + waveM * 16 + lrow;
  v16bf aq[2];
  aq[0] = make_afrag(qbase + (size_t)qrow * DHEAD + lhalf * 8);
  aq[1] = make_afrag(qbase + (size_t)qrow * DHEAD + 32 + lhalf * 8);

  v8f zero = {0.f, 0.f, 0.f, 0.f, 0.f, 0.f, 0.f, 0.f};
  v8f num[3] = {zero, zero, zero};

  // constant ones-column B fragment (col 64 -> 1.0, cols 65..79 -> 0)
  v16s os = {};
  if (lrow == 0) {
#pragma unroll
    for (int i = 0; i < 16; ++i) os[i] = (short)0x3F80;
  }
  v16bf onesFrag = __builtin_bit_cast(v16bf, os);

  for (int kt = 0; kt < TSEQ / 64; ++kt) {
    int key0 = kt * 64;
    // ---- S = q @ k^T, transform, stash bf16 scores in LDS ----
#pragma unroll
    for (int nt = 0; nt < 2; ++nt) {
      int colk = key0 + waveN * 32 + nt * 16 + lrow;  // global key index
      v8f sacc = zero;
#pragma unroll
      for (int j = 0; j < 2; ++j) {
        v16bf bk =
            make_bfrag(kbase + (size_t)colk * DHEAD + j * 32 + lhalf * 8);
        sacc = wmma_bf16(aq[j], bk, sacc);
      }
      float mv = (float)maskp[b * TSEQ + colk];
      int klocal = waveN * 32 + nt * 16 + lrow;
#pragma unroll
      for (int v = 0; v < 8; ++v) {
        float s = sacc[v] + 1.0f;
        float a = s * s * mv;
        int mlocal = waveM * 16 + lhalf * 8 + v;
        Asc[mlocal * 80 + klocal] = f2bf(a);
      }
    }
    __syncthreads();
    // ---- num += scores @ [V | 1] ----
#pragma unroll
    for (int j = 0; j < 2; ++j) {
      int mloc = waveM * 16 + lrow;
      v16bf a2 = make_afrag(&Asc[mloc * 80 + j * 32 + lhalf * 8]);
#pragma unroll
      for (int i = 0; i < 3; ++i) {
        int nt2 = waveN * 3 + i;  // waveN0: 0,1,2 ; waveN1: 3,4,(5 unused)
        v16bf b2;
        if (nt2 < 4) {
          int col = nt2 * 16 + lrow;  // output channel d
          b2 = make_bfrag(vbase + (size_t)col * TSEQ + key0 + j * 32 +
                          lhalf * 8);
        } else {
          b2 = onesFrag;  // denominator column (harmless if nt2==5)
        }
        num[i] = wmma_bf16(a2, b2, num[i]);
      }
    }
    __syncthreads();
  }

  // denominator lives in tile nt2==4 (waveN==1, i==1), column 64 (lrow==0)
  if (waveN == 1 && lrow == 0) {
#pragma unroll
    for (int v = 0; v < 8; ++v) dens[waveM * 16 + lhalf * 8 + v] = num[1][v];
  }
  __syncthreads();

#pragma unroll
  for (int i = 0; i < 3; ++i) {
    int nt2 = waveN * 3 + i;
    if (nt2 >= 4) continue;
    int col = nt2 * 16 + lrow;  // d in [0,64)
#pragma unroll
    for (int v = 0; v < 8; ++v) {
      int mloc = waveM * 16 + lhalf * 8 + v;
      float o = num[i][v] / (dens[mloc] + 1e-6f);
      int row = q0 + mloc;
      ao[((size_t)(b * TSEQ + row)) * HIDDEN + hh * DHEAD + col] = f2bf(o);
    }
  }
}

// ---------------------------------------------------------------------------
// Host launcher
// ---------------------------------------------------------------------------
extern "C" void kernel_launch(void* const* d_in, const int* in_sizes, int n_in,
                              void* d_out, int out_size, void* d_ws,
                              size_t ws_size, hipStream_t stream) {
  (void)in_sizes; (void)n_in; (void)out_size; (void)ws_size;
  const float* x     = (const float*)d_in[0];
  const int*   amask = (const int*)d_in[1];
  const float* ln1_g = (const float*)d_in[2];
  const float* ln1_b = (const float*)d_in[3];
  const float* ln2_g = (const float*)d_in[4];
  const float* ln2_b = (const float*)d_in[5];
  const float* w_qkv = (const float*)d_in[6];
  const float* w_o   = (const float*)d_in[7];
  const float* fc1_w = (const float*)d_in[8];
  const float* fc1_b = (const float*)d_in[9];
  const float* fc2_w = (const float*)d_in[10];
  const float* fc2_b = (const float*)d_in[11];
  const float* poly  = (const float*)d_in[12];

  unsigned short* qbuf  = (unsigned short*)d_ws;
  unsigned short* kbuf  = qbuf + (size_t)QSZ;
  unsigned short* vtbuf = kbuf + (size_t)QSZ;
  unsigned short* hbuf  = vtbuf + (size_t)QSZ;           // LN out (reused)
  unsigned short* aobuf = hbuf + (size_t)MTOT * HIDDEN;  // attention out
  float*          x1    = (float*)(aobuf + (size_t)MTOT * HIDDEN);
  unsigned short* ubuf  = (unsigned short*)(x1 + (size_t)MTOT * HIDDEN);
  unsigned short* wqkvT = ubuf + (size_t)MTOT * INTER;   // [2304,768]
  unsigned short* woT   = wqkvT + (size_t)3 * HIDDEN * HIDDEN;  // [768,768]
  unsigned short* fc1T  = woT + (size_t)HIDDEN * HIDDEN;        // [3072,768]
  unsigned short* fc2T  = fc1T + (size_t)INTER * HIDDEN;        // [768,3072]

  // 0. one-time weight transpose + bf16 convert (all independent)
  convert_wT<<<dim3(3 * HIDDEN / 32, HIDDEN / 32), 256, 0, stream>>>(
      w_qkv, wqkvT, HIDDEN, 3 * HIDDEN);
  convert_wT<<<dim3(HIDDEN / 32, HIDDEN / 32), 256, 0, stream>>>(
      w_o, woT, HIDDEN, HIDDEN);
  convert_wT<<<dim3(INTER / 32, HIDDEN / 32), 256, 0, stream>>>(
      fc1_w, fc1T, HIDDEN, INTER);
  convert_wT<<<dim3(HIDDEN / 32, INTER / 32), 256, 0, stream>>>(
      fc2_w, fc2T, INTER, HIDDEN);

  // 1. LN1
  ln_kernel<<<MTOT, 256, 0, stream>>>(x, ln1_g, ln1_b, hbuf);
  // 2. QKV projection + scatter
  gemm_kernel<0><<<dim3(3 * HIDDEN / 128, MTOT / 128), 256, 0, stream>>>(
      hbuf, wqkvT, 3 * HIDDEN, HIDDEN, nullptr, nullptr, nullptr, qbuf, nullptr);
  // 3. quadratic attention
  attn_kernel<<<dim3(TSEQ / 64, BATCH * HEADS), 256, 0, stream>>>(
      qbuf, kbuf, vtbuf, amask, aobuf);
  // 4. output projection + residual -> x1
  gemm_kernel<1><<<dim3(HIDDEN / 128, MTOT / 128), 256, 0, stream>>>(
      aobuf, woT, HIDDEN, HIDDEN, nullptr, x, nullptr, nullptr, x1);
  // 5. LN2
  ln_kernel<<<MTOT, 256, 0, stream>>>(x1, ln2_g, ln2_b, hbuf);
  // 6. FC1 + bias + poly-GELU
  gemm_kernel<2><<<dim3(INTER / 128, MTOT / 128), 256, 0, stream>>>(
      hbuf, fc1T, INTER, HIDDEN, fc1_b, nullptr, poly, ubuf, nullptr);
  // 7. FC2 + bias + residual -> d_out
  gemm_kernel<3><<<dim3(HIDDEN / 128, MTOT / 128), 256, 0, stream>>>(
      ubuf, fc2T, HIDDEN, INTER, fc2_b, x1, nullptr, nullptr, (float*)d_out);
}
